// MinkHead_64707977281696
// MI455X (gfx1250) — compile-verified, matrix operand order checked by reading, not run on hardware
//
#include <hip/hip_runtime.h>

// MI455X / gfx1250 — wave32 WMMA bf16 path with async global->LDS staging.
//
// FPN merge in scatter-GEMM form (all shapes N=128):
//   Y3 = feats3 @ W3                      (K=256)
//   Z3[m,k,:] = Y3[m,:] @ T3[k]           (K=128, grid.y = 8)
//   Y2 = Z3[parent2, offset2] + feats2@W2 (K=128, gather fused in epilogue)
//   Z2[m,k,:] = Y2[m,:] @ T2[k]           (K=128, grid.y = 8)
//   out = Z2[parent1, offset1] + feats1@W1(K=64,  gather fused)
//
// Roofline: ~52 GFLOP vs ~1.3 GB HBM -> bandwidth bound (~56 us floor at
// 23.3 TB/s) provided matrix math runs on v_wmma_f32_16x16x32_bf16.
//
// Structure per block (256 threads = 8 waves, 32-row M tile):
//  - B fragments for the whole K live in VGPRs (<=64 VGPRs at K=256);
//    weights are tiny and L2-resident, loaded once per block.
//  - A tile (32x32 f32 = 4 KB) staged to LDS: exactly one
//    global_load_async_to_lds_b128 per thread per K-chunk, double buffered,
//    synced with s_wait_asynccnt + workgroup barrier.
//  - Per K-chunk: 2 WMMAs (rows 0-15 / 16-31) sharing one register-resident
//    B fragment; K is a template param so everything fully unrolls.

#define GLOBAL_AS __attribute__((address_space(1)))
#define LDS_AS    __attribute__((address_space(3)))

#if defined(__has_builtin)
#  if __has_builtin(__builtin_amdgcn_global_load_async_to_lds_b128) && \
      __has_builtin(__builtin_amdgcn_s_wait_asynccnt)
#    define USE_ASYNC_LDS 1
#  endif
#endif
#ifndef USE_ASYNC_LDS
#  define USE_ASYNC_LDS 0
#endif

typedef __attribute__((ext_vector_type(4)))  int    v4i;
typedef __attribute__((ext_vector_type(16))) __bf16 v16bf;
typedef __attribute__((ext_vector_type(8)))  float  v8f;

#define KC 32          // K per WMMA (bf16: 16x16x32)
#define MT 32          // M rows per block (2 WMMA row-tiles)
#define AS_STRIDE 36   // f32 row stride: 144B (16B aligned), rows land on
                       // distinct 4-bank blocks (36*r mod 64 has period 16)

// CDNA5 16-bit fragment K pattern (CDNA5 ISA §7.12.2):
// lane<16: K = {0..7, 16..23}; lane>=16: +8. Element e -> VGPR e/2, half e&1.
__device__ __forceinline__ int frag_k(int e, int hi) {
  int v = e >> 1;
  return ((v & 4) << 2) + (hi << 3) + ((v & 3) << 1) + (e & 1);
}

__device__ __forceinline__ v8f wmma_bf16(v16bf a, v16bf b, v8f c) {
  return __builtin_amdgcn_wmma_f32_16x16x32_bf16(
      /*neg_a=*/false, a, /*neg_b=*/false, b,
      /*c_mod=*/(short)0, c, /*reuse_a=*/false, /*reuse_b=*/false);
}

// Stage one 32xKC f32 A-chunk into an LDS buffer: one b128 per thread.
template <int K>
__device__ __forceinline__ void stage_A(const float* __restrict__ A,
                                        float* __restrict__ dstBuf,
                                        int tid, int row0, int kc, int M) {
  const int row = tid >> 3;          // 0..31
  const int c4  = (tid & 7) << 2;    // 0,4,...,28
  int gr = row0 + row;
  if (gr > M - 1) gr = M - 1;        // clamp tail rows (dupes, stores guarded)
  const float* src = A + (size_t)gr * K + (kc + c4);
  float*       dst = dstBuf + row * AS_STRIDE + c4;
#if USE_ASYNC_LDS
  __builtin_amdgcn_global_load_async_to_lds_b128(
      (GLOBAL_AS v4i*)src, (LDS_AS v4i*)dst, /*offset=*/0, /*cpol=*/0);
#else
  *(float4*)dst = *(const float4*)src;
#endif
}

__device__ __forceinline__ void async_join() {
#if USE_ASYNC_LDS
  __builtin_amdgcn_s_wait_asynccnt(0);
#endif
  __syncthreads();
}

// Read one 16x32 bf16 A fragment from a staged f32 tile: 8x ds_load_b64
// (K-pairs are contiguous) + packed f32->bf16 converts.
__device__ __forceinline__ v16bf read_afrag(const float* __restrict__ buf,
                                            int lr, int hi) {
  v16bf a;
#pragma unroll
  for (int v = 0; v < 8; ++v) {
    const int k = ((v & 4) << 2) + (hi << 3) + ((v & 3) << 1);
    float2 f = *(const float2*)&buf[lr * AS_STRIDE + k];
    a[2 * v]     = (__bf16)f.x;
    a[2 * v + 1] = (__bf16)f.y;
  }
  return a;
}

template <int K, bool GATHER>
__global__ __launch_bounds__(256, 2)
void fpn_wmma_gemm(const float* __restrict__ A,       // [M x K] row-major
                   const float* __restrict__ B,       // [gridDim.y][K x 128]
                   float*       __restrict__ out,     // [M x gridDim.y x 128]
                   const float* __restrict__ gatherZ, // [Mc x 8 x 128] or null
                   const int*   __restrict__ parent,  // [M] or null
                   const int*   __restrict__ offset,  // [M] or null
                   int M) {
  constexpr int NKC = K / KC;
  __shared__ float As[2][MT * AS_STRIDE];

  const int tid  = threadIdx.x;
  const int wave = tid >> 5;          // 8 waves x 16 cols = 128 = D
  const int lane = tid & 31;
  const int lr   = lane & 15;
  const int hi   = lane >> 4;
  const int row0 = blockIdx.x * MT;
  const int col  = wave * 16 + lr;
  const float* __restrict__ Bp = B + (size_t)blockIdx.y * (K * 128);

  // Preload ALL B fragments into registers (B: col = lane%16, K pattern as A).
  v16bf bfrag[NKC];
#pragma unroll
  for (int c = 0; c < NKC; ++c) {
#pragma unroll
    for (int e = 0; e < 16; ++e) {
      const int k = frag_k(e, hi);
      bfrag[c][e] = (__bf16)Bp[(size_t)(c * KC + k) * 128 + col];
    }
  }

  v8f acc0 = {};  // rows row0 +  0..15
  v8f acc1 = {};  // rows row0 + 16..31

  stage_A<K>(A, As[0], tid, row0, 0, M);
  async_join();

#pragma unroll
  for (int c = 0; c < NKC; ++c) {
    const float* buf = As[c & 1];
    if (c + 1 < NKC)  // issue next chunk's async DMA into the other buffer
      stage_A<K>(A, As[(c + 1) & 1], tid, row0, (c + 1) * KC, M);

    v16bf a0 = read_afrag(buf, lr, hi);
    v16bf a1 = read_afrag(buf + 16 * AS_STRIDE, lr, hi);
    acc0 = wmma_bf16(a0, bfrag[c], acc0);
    acc1 = wmma_bf16(a1, bfrag[c], acc1);

    if (c + 1 < NKC) async_join();  // DMA landed + all waves done reading buf
  }

  // Epilogue. C/D layout: VGPR r holds row (r + 8*hi), col = wave*16 + lane%16.
  const size_t OB = gridDim.y;
#pragma unroll
  for (int r = 0; r < 8; ++r) {
    const int m0 = row0 + r + (hi << 3);
    const int m1 = m0 + 16;
    float v0 = acc0[r];
    float v1 = acc1[r];
    if (m0 < M) {
      if (GATHER)
        v0 += gatherZ[(((size_t)parent[m0] << 3) + (size_t)offset[m0]) * 128 + col];
      out[((size_t)m0 * OB + blockIdx.y) * 128 + col] = v0;
    }
    if (m1 < M) {
      if (GATHER)
        v1 += gatherZ[(((size_t)parent[m1] << 3) + (size_t)offset[m1]) * 128 + col];
      out[((size_t)m1 * OB + blockIdx.y) * 128 + col] = v1;
    }
  }
}

extern "C" void kernel_launch(void* const* d_in, const int* in_sizes, int n_in,
                              void* d_out, int out_size, void* d_ws, size_t ws_size,
                              hipStream_t stream) {
  const float* feats1  = (const float*)d_in[0];
  const float* feats2  = (const float*)d_in[1];
  const float* feats3  = (const float*)d_in[2];
  const int*   parent1 = (const int*)  d_in[3];
  const int*   offset1 = (const int*)  d_in[4];
  const int*   parent2 = (const int*)  d_in[5];
  const int*   offset2 = (const int*)  d_in[6];
  const float* W1      = (const float*)d_in[7];
  const float* W2      = (const float*)d_in[8];
  const float* W3      = (const float*)d_in[9];
  const float* T2      = (const float*)d_in[10];
  const float* T3      = (const float*)d_in[11];

  const int D  = 128;
  const int KV = 8;
  const int C1 = in_sizes[7] / D;   // 64
  const int C2 = in_sizes[8] / D;   // 128
  const int C3 = in_sizes[9] / D;   // 256
  const int N1 = in_sizes[0] / C1;  // 400000
  const int N2 = in_sizes[1] / C2;  // 120000
  const int N3 = in_sizes[2] / C3;  // 30000

  // Workspace (fp32): Y3 | Z3 | Y2 | Z2  (~691 MB total)
  float* ws = (float*)d_ws;
  float* Y3 = ws;                    size_t o = (size_t)N3 * D;
  float* Z3 = ws + o;                o += (size_t)N3 * KV * D;
  float* Y2 = ws + o;                o += (size_t)N2 * D;
  float* Z2 = ws + o;                // needs (o + N2*KV*D)*4 bytes <= ws_size
  (void)ws_size; (void)n_in; (void)out_size;

  dim3 blk(256);
  auto gx = [](int M) { return dim3((unsigned)((M + MT - 1) / MT), 1); };
  auto gxy = [KV](int M) { return dim3((unsigned)((M + MT - 1) / MT), (unsigned)KV); };

  // 1) Y3 = feats3 @ W3                        (K=256)
  fpn_wmma_gemm<256, false><<<gx(N3), blk, 0, stream>>>(
      feats3, W3, Y3, nullptr, nullptr, nullptr, N3);

  // 2) Z3[m,k,:] = Y3 @ T3[k]                  (K=128, 8 GEMMs via grid.y)
  fpn_wmma_gemm<128, false><<<gxy(N3), blk, 0, stream>>>(
      Y3, T3, Z3, nullptr, nullptr, nullptr, N3);

  // 3) Y2 = gather(Z3; parent2, offset2) + feats2 @ W2   (K=128)
  fpn_wmma_gemm<128, true><<<gx(N2), blk, 0, stream>>>(
      feats2, W2, Y2, Z3, parent2, offset2, N2);

  // 4) Z2[m,k,:] = Y2 @ T2[k]                  (K=128)
  fpn_wmma_gemm<128, false><<<gxy(N2), blk, 0, stream>>>(
      Y2, T2, Z2, nullptr, nullptr, nullptr, N2);

  // 5) out = gather(Z2; parent1, offset1) + feats1 @ W1  (K=64)
  fpn_wmma_gemm<64, true><<<gx(N1), blk, 0, stream>>>(
      feats1, W1, (float*)d_out, Z2, parent1, offset1, N1);
}
